// EmbeddingNet_80556406604203
// MI455X (gfx1250) — matrix operand order repaired and checked by reading, not compiled, not run on hardware
//
#include <hip/hip_runtime.h>
#include <hip/hip_bf16.h>
#include <cmath>

// ---------------------------------------------------------------------------
// MI455X (gfx1250): wave32, WMMA 16x16x32 bf16 (f32 accumulate).
// B=131072, IN=256, D=64. e_issue dead in reference -> skipped.
// Compute-bound (~160 GFLOP vs ~400 MB HBM). Changes vs last round:
//  * weights pre-converted to bf16 once (kills per-loop v_cvt VALU pressure)
//  * per-block LDS staging of w1/w2 (bf16) via TDM tensor_load_to_lds when
//    available (TENSORcnt-tracked), cooperative copy fallback
//  * dynamic LDS: 64KB h1 + 192KB weight buffer = 256KB of 320KB/WGP
// ---------------------------------------------------------------------------

typedef __bf16 bf16_t;
typedef __attribute__((ext_vector_type(16))) __bf16 v16bf;
typedef __attribute__((ext_vector_type(8)))  __bf16 v8bf;
typedef __attribute__((ext_vector_type(8)))  float  v8f;
typedef __attribute__((ext_vector_type(4)))  unsigned int u32x4;
typedef __attribute__((ext_vector_type(8)))  int i32x8;
typedef __attribute__((ext_vector_type(4)))  int i32x4;

#define BATCH  131072
#define IN_DIM 256
#define EMB_D  64

#if defined(__has_builtin)
#if __has_builtin(__builtin_amdgcn_tensor_load_to_lds) && \
    __has_builtin(__builtin_amdgcn_s_wait_tensorcnt)
#define HAVE_TDM 1
#endif
#endif
#ifndef HAVE_TDM
#define HAVE_TDM 0
#endif

static __device__ inline v8f wmma_bf16(v16bf a, v16bf b, v8f c) {
  return __builtin_amdgcn_wmma_f32_16x16x32_bf16(
      /*neg_a=*/false, a, /*neg_b=*/false, b,
      /*c_mod=*/(short)0, c, /*reuse_a=*/false, /*reuse_b=*/false);
}

// A fragment (16x32) from an fp32 row; koff already includes lane-half offset.
static __device__ inline v16bf load_a_f32(const float* row, int koff) {
  v16bf a;
#pragma unroll
  for (int i = 0; i < 8; ++i) a[i]     = (__bf16)row[koff + i];
#pragma unroll
  for (int i = 0; i < 8; ++i) a[8 + i] = (__bf16)row[koff + 16 + i];
  return a;
}

// A fragment from a bf16 row (global or LDS): K {koff..+7, koff+16..+23}.
static __device__ inline v16bf load_a_bf16(const bf16_t* row, int koff) {
  union { v16bf v; v8bf h[2]; } u;
  u.h[0] = *(const v8bf*)(row + koff);
  u.h[1] = *(const v8bf*)(row + koff + 16);
  return u.v;
}

// B fragment (32x16) for D = X @ W^T from bf16 W row-major [N,K]:
// lane n = lane&15; K {kbase..+7, kbase+16..+23} (kbase has +8 for hi half).
static __device__ inline v16bf load_b_bf16(const bf16_t* w, int ld, int n, int kbase) {
  const bf16_t* p = w + (size_t)n * ld + kbase;
  union { v16bf v; v8bf h[2]; } u;
  u.h[0] = *(const v8bf*)(p);
  u.h[1] = *(const v8bf*)(p + 16);
  return u.v;
}

// ---------------------------------------------------------------------------
// Kernel 0: one-shot fp32 -> bf16 weight conversion (11 matrices, ~1.2 MB)
// ---------------------------------------------------------------------------
struct CvtArgs {
  const float* src[11];
  bf16_t*      dst[11];
  int          n[11];
};

__global__ __launch_bounds__(256) void cvt_kernel(CvtArgs a) {
  const int j = blockIdx.y;
  const float* s = a.src[j];
  bf16_t*      d = a.dst[j];
  const int    n = a.n[j];
  for (int i = blockIdx.x * blockDim.x + threadIdx.x; i < n;
       i += gridDim.x * blockDim.x)
    d[i] = (bf16_t)s[i];
}

// ---------------------------------------------------------------------------
// LDS staging: TDM (tensor_load_to_lds, TENSORcnt) or cooperative copy.
// ---------------------------------------------------------------------------
#if HAVE_TDM
static __device__ inline void tdm_stage(bf16_t* dst_lds, const bf16_t* src,
                                        unsigned nbytes) {
  const unsigned ndw = nbytes >> 2;                     // 4-byte elements
  const unsigned lds_off = (unsigned)(uintptr_t)dst_lds; // generic lo32 = LDS off
  const unsigned long long ga = (unsigned long long)(uintptr_t)src;
  u32x4 g0;
  g0[0] = 1u;                                           // count=1, user desc
  g0[1] = lds_off;                                      // lds_addr
  g0[2] = (unsigned)(ga & 0xFFFFFFFFu);                 // global_addr[31:0]
  g0[3] = (unsigned)((ga >> 32) & 0x1FFFFFFu) | (2u << 30); // [56:32] | type=2
  i32x8 g1;
  g1[0] = (int)(2u << 16);                              // data_size = 4B
  g1[1] = (int)((ndw & 0xFFFFu) << 16);                 // tensor_dim0[15:0]
  g1[2] = (int)(((ndw >> 16) & 0xFFFFu) | (1u << 16));  // dim0[31:16] | dim1=1
  g1[3] = (int)((ndw & 0xFFFFu) << 16);                 // dim1 hi=0 | tile_dim0
  g1[4] = 1;                                            // tile_dim1=1, dim2=0
  g1[5] = (int)ndw;                                     // dim0_stride[31:0]
  g1[6] = 0;                                            // stride0 hi | stride1 lo
  g1[7] = 0;
  i32x4 z4 = {0, 0, 0, 0};
#if __clang_major__ >= 23
  i32x8 z8 = {0, 0, 0, 0, 0, 0, 0, 0};
  __builtin_amdgcn_tensor_load_to_lds(g0, g1, z4, z4, z8, 0);
#else
  __builtin_amdgcn_tensor_load_to_lds(g0, g1, z4, z4, 0);
#endif
}
#endif

// Stage nbytes (16B multiple) of bf16 weights into LDS; all threads call,
// ends fully synchronized.
static __device__ inline void stage_weights(bf16_t* dst_lds, const bf16_t* src,
                                            unsigned nbytes) {
#if HAVE_TDM
  if (threadIdx.x < 32) {               // one wave issues the DMA
    tdm_stage(dst_lds, src, nbytes);
    __builtin_amdgcn_s_wait_tensorcnt(0);
  }
  __syncthreads();
#else
  const uint4* s = (const uint4*)src;
  uint4* d = (uint4*)dst_lds;
  const int n = nbytes >> 4;
  for (int i = threadIdx.x; i < n; i += 128) d[i] = s[i];
  __syncthreads();
#endif
}

// ---------------------------------------------------------------------------
// Kernel 1: entity embeddings  e = x @ w^T + b  -> bf16 [B,64]
// grid = (B/128, 3), block = 256 (8 waves x 16 rows); weights already bf16.
// ---------------------------------------------------------------------------
struct EmbArgs {
  const float*  x[3];
  const bf16_t* wq[3];
  const float*  b[3];
  bf16_t*       out[3];
};

__global__ __launch_bounds__(256) void emb_kernel(EmbArgs args) {
  const int ent = blockIdx.y;
  const float*  x    = args.x[ent];
  const bf16_t* wq   = args.wq[ent];
  const float*  bias = args.b[ent];
  bf16_t*       out  = args.out[ent];

  const int lane    = threadIdx.x & 31;
  const int wave    = threadIdx.x >> 5;
  const int rowbase = (blockIdx.x * 8 + wave) * 16;

  const int am    = rowbase + (lane & 15);
  const int akoff = (lane >> 4) ? 8 : 0;
  const float* xrow = x + (size_t)am * IN_DIM;

  v16bf afr[8];                         // K=256 -> 8 tiles of 32
#pragma unroll
  for (int kt = 0; kt < 8; ++kt) afr[kt] = load_a_f32(xrow, kt * 32 + akoff);

  const int bn     = lane & 15;
  const int bkhalf = (lane >> 4) ? 8 : 0;
  const int mstore = rowbase + ((lane >> 4) ? 8 : 0);

#pragma unroll
  for (int nt = 0; nt < 4; ++nt) {
    v8f acc = {};
#pragma unroll
    for (int kt = 0; kt < 8; ++kt) {
      v16bf bfr = load_b_bf16(wq, IN_DIM, nt * 16 + bn, kt * 32 + bkhalf);
      acc = wmma_bf16(afr[kt], bfr, acc);
    }
    const int n = nt * 16 + bn;
    const float bv = bias[n];
#pragma unroll
    for (int r = 0; r < 8; ++r)
      out[(size_t)(mstore + r) * EMB_D + n] = (bf16_t)(acc[r] + bv);
  }
}

// ---------------------------------------------------------------------------
// Kernel 2: 4 task MLPs per 16-row strip + per-row task selection
// grid = B/64, block = 128 (4 waves); dynamic LDS = 64KB h1 + 192KB weights
// ---------------------------------------------------------------------------
struct MlpArgs {
  const bf16_t* e[3];                   // c, r, p embeddings [B,64] bf16
  const bf16_t* w1q[4];                 // bf16 [512, Dk]
  const bf16_t* w2q[4];                 // bf16 [128, 512]
  const float*  b1[4];
  const float*  b2[4];
  const float*  w3[4];
  const float*  b3[4];
  const float*  task_one_hot;           // [B,4]
  float*        out;                    // [B]
};

template <int T>
__device__ inline void run_task(const v16bf (&efrag)[6], bf16_t* h1w,
                                bf16_t* wbuf, const MlpArgs& args,
                                int lane, int rowbase) {
  // ktile -> embedding fragment (c=0/1, r=2/3, p=4/5), per ALL_TASK_ENTS
  constexpr int FM[4][6] = {{0,1,2,3,0,0},   // t0: [c,r]
                            {2,3,4,5,0,1},   // t1: [r,p,c]
                            {2,3,0,1,0,0},   // t2: [r,c]
                            {0,1,0,1,0,0}};  // t3: [c,c]
  constexpr int NKT = (T == 1) ? 6 : 4;
  constexpr int DK  = (T == 1) ? 192 : 128;

  const float* b1p = args.b1[T];
  const float* b2p = args.b2[T];
  const float* w3p = args.w3[T];
  const float  b3v = args.b3[T][0];

  const int bn     = lane & 15;
  const int bkhalf = (lane >> 4) ? 8 : 0;
  const int akoff  = (lane >> 4) ? 8 : 0;
  const int mhalf  = (lane >> 4) ? 8 : 0;

  // ---- stage w1 (bf16) into LDS, then Layer 1 -> h1 (per-wave LDS) ----
  __syncthreads();                       // prior task done with wbuf
  stage_weights(wbuf, args.w1q[T], 512u * DK * 2u);

  for (int nt = 0; nt < 32; ++nt) {      // N = 512
    v8f acc = {};
#pragma unroll
    for (int kt = 0; kt < NKT; ++kt) {
      v16bf bfr = load_b_bf16(wbuf, DK, nt * 16 + bn, kt * 32 + bkhalf);
      acc = wmma_bf16(efrag[FM[T][kt]], bfr, acc);
    }
    const int n = nt * 16 + bn;
    const float bv = b1p[n];
#pragma unroll
    for (int r = 0; r < 8; ++r) {
      float v = acc[r] + bv;
      h1w[(mhalf + r) * 512 + n] = (bf16_t)(v > 0.f ? v : 0.f);
    }
  }

  // ---- stage w2, then Layer 2 + fused layer-3 partials ----
  __syncthreads();                       // all waves done reading w1 in wbuf
  stage_weights(wbuf, args.w2q[T], 128u * 512u * 2u);

  float p3[8];
#pragma unroll
  for (int r = 0; r < 8; ++r) p3[r] = 0.f;

  const bf16_t* arow = h1w + (lane & 15) * 512;
#pragma unroll
  for (int nt = 0; nt < 8; ++nt) {       // N = 128
    v8f acc = {};
#pragma unroll
    for (int kt = 0; kt < 16; ++kt) {    // K = 512
      v16bf afr = load_a_bf16(arow, kt * 32 + akoff);
      v16bf bfr = load_b_bf16(wbuf, 512, nt * 16 + bn, kt * 32 + bkhalf);
      acc = wmma_bf16(afr, bfr, acc);
    }
    const int n = nt * 16 + bn;
    const float bv  = b2p[n];
    const float w3v = w3p[n];
#pragma unroll
    for (int r = 0; r < 8; ++r) {
      float h = acc[r] + bv;
      h = h > 0.f ? h : 0.f;
      p3[r] += h * w3v;
    }
  }

  // ---- 16-lane reduce, sigmoid, per-row task select, store ----
#pragma unroll
  for (int r = 0; r < 8; ++r) {
    float s = p3[r];
    s += __shfl_xor(s, 1, 32);
    s += __shfl_xor(s, 2, 32);
    s += __shfl_xor(s, 4, 32);
    s += __shfl_xor(s, 8, 32);
    if ((lane & 15) == 0) {
      const int gm = rowbase + r + mhalf;
      const float* oh = args.task_one_hot + (size_t)gm * 4;
      int idx = 0; float best = oh[0];
#pragma unroll
      for (int j = 1; j < 4; ++j) {
        float v = oh[j];
        if (v > best) { best = v; idx = j; }
      }
      if (idx == T) {
        const float logit = s + b3v;
        args.out[gm] = 1.f / (1.f + expf(-logit));
      }
    }
  }
}

__global__ __launch_bounds__(128) void mlp_kernel(MlpArgs args) {
  extern __shared__ char smem_raw[];
  bf16_t* h1base = (bf16_t*)smem_raw;                     // 4*16*512 = 64KB
  bf16_t* wbuf   = (bf16_t*)(smem_raw + 4 * 16 * 512 * 2); // up to 192KB

  const int lane    = threadIdx.x & 31;
  const int wave    = threadIdx.x >> 5;
  const int rowbase = (blockIdx.x * 4 + wave) * 16;

  const int am    = rowbase + (lane & 15);
  const int akoff = (lane >> 4) ? 8 : 0;

  v16bf efrag[6];                        // c,r,p x two K=32 tiles
#pragma unroll
  for (int ent = 0; ent < 3; ++ent) {
    const bf16_t* row = args.e[ent] + (size_t)am * EMB_D;
    efrag[ent * 2 + 0] = load_a_bf16(row, 0  + akoff);
    efrag[ent * 2 + 1] = load_a_bf16(row, 32 + akoff);
  }

  bf16_t* h1w = h1base + wave * 16 * 512;
  run_task<0>(efrag, h1w, wbuf, args, lane, rowbase);
  run_task<1>(efrag, h1w, wbuf, args, lane, rowbase);
  run_task<2>(efrag, h1w, wbuf, args, lane, rowbase);
  run_task<3>(efrag, h1w, wbuf, args, lane, rowbase);
}

// ---------------------------------------------------------------------------
extern "C" void kernel_launch(void* const* d_in, const int* in_sizes, int n_in,
                              void* d_out, int out_size, void* d_ws, size_t ws_size,
                              hipStream_t stream) {
  const float* task_oh = (const float*)d_in[0];
  const float* c_x = (const float*)d_in[1];
  const float* r_x = (const float*)d_in[2];
  const float* p_x = (const float*)d_in[3];
  // d_in[4] (i_x), d_in[11..12] (w_i/b_i): issue embedding unused.
  const float* w_ent[3] = { (const float*)d_in[5], (const float*)d_in[7],
                            (const float*)d_in[9] };
  const float* b_ent[3] = { (const float*)d_in[6], (const float*)d_in[8],
                            (const float*)d_in[10] };

  const int dk[4] = {128, 192, 128, 128};

  // ---- workspace layout: e_c/e_r/e_p (bf16) then bf16 weights ----
  bf16_t* e_base = (bf16_t*)d_ws;                 // 3 * B * 64 * 2 = 50.3 MB
  const size_t estride = (size_t)BATCH * EMB_D;
  bf16_t* wp = e_base + 3 * estride;

  CvtArgs ca;
  bf16_t* wq_emb[3];
  for (int i = 0; i < 3; ++i) {
    wq_emb[i] = wp;
    ca.src[i] = w_ent[i]; ca.dst[i] = wp; ca.n[i] = EMB_D * IN_DIM;
    wp += EMB_D * IN_DIM;
  }
  bf16_t* wq1[4];
  for (int t = 0; t < 4; ++t) {
    wq1[t] = wp;
    ca.src[3 + t] = (const float*)d_in[13 + 6 * t];
    ca.dst[3 + t] = wp; ca.n[3 + t] = 512 * dk[t];
    wp += 512 * dk[t];
  }
  bf16_t* wq2[4];
  for (int t = 0; t < 4; ++t) {
    wq2[t] = wp;
    ca.src[7 + t] = (const float*)d_in[15 + 6 * t];
    ca.dst[7 + t] = wp; ca.n[7 + t] = 128 * 512;
    wp += 128 * 512;
  }
  cvt_kernel<<<dim3(64, 11), 256, 0, stream>>>(ca);

  EmbArgs ea;
  for (int i = 0; i < 3; ++i) {
    ea.x[i] = (i == 0) ? c_x : (i == 1) ? r_x : p_x;
    ea.wq[i] = wq_emb[i];
    ea.b[i] = b_ent[i];
    ea.out[i] = e_base + i * estride;
  }
  emb_kernel<<<dim3(BATCH / 128, 3), 256, 0, stream>>>(ea);

  MlpArgs ma;
  for (int i = 0; i < 3; ++i) ma.e[i] = ea.out[i];
  for (int t = 0; t < 4; ++t) {
    ma.w1q[t] = wq1[t];
    ma.w2q[t] = wq2[t];
    ma.b1[t] = (const float*)d_in[14 + 6 * t];
    ma.b2[t] = (const float*)d_in[16 + 6 * t];
    ma.w3[t] = (const float*)d_in[17 + 6 * t];
    ma.b3[t] = (const float*)d_in[18 + 6 * t];
  }
  ma.task_one_hot = task_oh;
  ma.out = (float*)d_out;

  // dynamic LDS: 64KB h1 + 192KB weight stage = 256KB (<= 320KB/WGP)
  mlp_kernel<<<dim3(BATCH / 64), 128, 262144, stream>>>(ma);
}
